// MultiHeadLocalAttention_63926293234199
// MI455X (gfx1250) — matrix operand order repaired and checked by reading, not compiled, not run on hardware
//
#include <hip/hip_runtime.h>
#include <hip/hip_bf16.h>
#include <stdint.h>

// ---------------- CDNA5 WMMA types ----------------
typedef __attribute__((ext_vector_type(16))) __bf16 v16bf;
typedef __attribute__((ext_vector_type(8)))  float  v8f;
typedef __attribute__((ext_vector_type(4)))  unsigned int v4u;
typedef int i32x4v __attribute__((vector_size(16)));   // matches builtin param pointee

#define BATCH 2
#define T_SEQ 2048
#define CDIM  1024
#define HEADS 16
#define DHEAD 64
#define WINSZ 256

#if __has_builtin(__builtin_amdgcn_global_load_async_to_lds_b128) && \
    __has_builtin(__builtin_amdgcn_s_wait_asynccnt)
#define HAVE_ASYNC_LDS 1
#else
#define HAVE_ASYNC_LDS 0
#endif

typedef __attribute__((address_space(1))) i32x4v* gptr_b128;
typedef __attribute__((address_space(3))) i32x4v* lptr_b128;

__device__ __forceinline__ unsigned short f2bf(float f) {
  unsigned u = __float_as_uint(f);
  u += 0x7fffu + ((u >> 16) & 1u);   // round-to-nearest-even
  return (unsigned short)(u >> 16);
}

union V16U {
  v4u u4[2];
  unsigned short us[16];
  v16bf v;
};

// A operand (16x32 bf16, M x K). ISA 7.12.2: lane L holds row M = L%16;
// lanes 0-15: e=0..7 -> K=e, e=8..15 -> K=e+8; lanes 16-31: +8.
__device__ __forceinline__ v16bf load_a16(const unsigned short* __restrict__ base,
                                          int ld, int row0, int k0, int lane) {
  const int half = (lane >> 4) & 1, r = lane & 15;
  const unsigned short* p = base + (size_t)(row0 + r) * ld + k0 + half * 8;
  V16U t;
  t.u4[0] = *(const v4u*)(p);        // K = k0 + half*8 + 0..7
  t.u4[1] = *(const v4u*)(p + 16);   // K = k0 + 16 + half*8 + 0..7
  return t.v;
}

// B operand (32x16 bf16, K x N). lane L holds col N = L%16;
// lanes 0-15: e=0..7 -> K=e, e=8..15 -> K=e+8; lanes 16-31: +8 (matches A).
__device__ __forceinline__ v16bf load_b16(const unsigned short* __restrict__ base,
                                          int ld, int col0, int k0, int lane) {
  const int half = (lane >> 4) & 1, r = lane & 15;
  const unsigned short* p = base + (size_t)(col0 + r) * ld + k0 + half * 8;
  V16U t;
  t.u4[0] = *(const v4u*)(p);
  t.u4[1] = *(const v4u*)(p + 16);
  return t.v;
}

// ---------------- kernels ----------------

__global__ __launch_bounds__(256) void cvt_bf16(const float* __restrict__ in,
                                                unsigned short* __restrict__ out, int n) {
  int i = blockIdx.x * 256 + threadIdx.x;
  if (i < n) out[i] = f2bf(in[i]);
}

// C[m,n] = sum_k A[m,k]*Bw[n,k] (+bias[n]); A,Bw bf16 row-major with leading dim K.
// One 16x16 tile per wave, 8 waves/block => 16x128 macro tile.
__global__ __launch_bounds__(256) void gemm_bf16_16x16(
    const unsigned short* __restrict__ A, const unsigned short* __restrict__ Bw,
    const float* __restrict__ bias, float* __restrict__ Cout,
    int N, int K, int nMacroCnt) {
  const int wave = threadIdx.x >> 5;
  const int lane = threadIdx.x & 31;
  const int mt = blockIdx.x / nMacroCnt;
  const int nm = blockIdx.x % nMacroCnt;
  const int m0 = mt * 16;
  const int n0 = nm * 128 + wave * 16;
  const int r = lane & 15;
  v8f acc = {0.f, 0.f, 0.f, 0.f, 0.f, 0.f, 0.f, 0.f};
  for (int k = 0; k < K; k += 32) {
    v16bf a = load_a16(A, K, m0, k, lane);
    v16bf b = load_b16(Bw, K, n0, k, lane);
    if (k + 256 < K) {   // hint the streaming path ahead (global_prefetch_b8)
      __builtin_prefetch(A + (size_t)(m0 + r) * K + k + 256, 0, 1);
      __builtin_prefetch(Bw + (size_t)(n0 + r) * K + k + 256, 0, 1);
    }
    acc = __builtin_amdgcn_wmma_f32_16x16x32_bf16(false, a, false, b, (short)0, acc,
                                                  false, false);
  }
  const int half = lane >> 4;
  const float bv = bias ? bias[n0 + r] : 0.f;
#pragma unroll
  for (int g = 0; g < 8; ++g) {
    const int mm = m0 + g + 8 * half;                 // C layout: VGPR g -> M = g + 8*half
    Cout[(size_t)mm * N + (n0 + r)] = acc[g] + bv;    // lane -> N
  }
}

// qkv fp32 (B,T,3C) -> RoPE(q,k), pack Q/K/V bf16 as (B,H,T,D). One thread per rot-pair.
__global__ __launch_bounds__(256) void rope_pack(
    const float* __restrict__ qkv,
    unsigned short* __restrict__ Qb, unsigned short* __restrict__ Kb,
    unsigned short* __restrict__ Vb) {
  const int tid = blockIdx.x * 256 + threadIdx.x;
  const int i  = tid & 31;                 // pair index (D/2 = 32)
  const int t  = (tid >> 5) & (T_SEQ - 1);
  const int bh = tid >> 16;                // tid / (32*T)
  const float* row = qkv + (size_t)((bh >> 4) * T_SEQ + t) * (3 * CDIM)
                         + (bh & (HEADS - 1)) * (3 * DHEAD);
  const float theta = __powf(10000.f, -(float)i * (1.f / 32.f));
  float s, c;
  __sincosf((float)t * theta, &s, &c);
  const float q1 = row[2 * i], q2 = row[2 * i + 1];
  const float k1 = row[DHEAD + 2 * i], k2 = row[DHEAD + 2 * i + 1];
  const float v1 = row[2 * DHEAD + 2 * i], v2 = row[2 * DHEAD + 2 * i + 1];
  const size_t o = ((size_t)bh * T_SEQ + t) * DHEAD + 2 * i;
  Qb[o]     = f2bf(q1 * c - q2 * s);
  Qb[o + 1] = f2bf(q1 * s + q2 * c);
  Kb[o]     = f2bf(k1 * c - k2 * s);
  Kb[o + 1] = f2bf(k1 * s + k2 * c);
  Vb[o]     = f2bf(v1);
  Vb[o + 1] = f2bf(v2);
}

// Local-window attention. One wave = one 16-query tile of one (b,h).
// 2 waves/block. LDS: fp32 score strip (16 x <=272) per wave + V chunk stage + stats.
__global__ __launch_bounds__(64) void attn_local(
    const unsigned short* __restrict__ Qb, const unsigned short* __restrict__ Kb,
    const unsigned short* __restrict__ Vb, unsigned short* __restrict__ Ob) {
  __shared__ float s_sc[2][16][289];
  __shared__ unsigned short s_v[2][32 * DHEAD];   // 32-key V chunk per wave (4KB)
  __shared__ float s_rm[2][16];
  __shared__ float s_rs[2][16];
  const int wave = threadIdx.x >> 5;
  const int lane = threadIdx.x & 31;
  const int half = lane >> 4, r = lane & 15;
  const int tile = blockIdx.x * 2 + wave;
  const int bh = tile >> 7;                 // tile / (T/16)
  const int qt = tile & 127;
  const int q0 = qt * 16;
  const size_t base = (size_t)bh * T_SEQ * DHEAD;
  const unsigned short* Qp = Qb + base;
  const unsigned short* Kp = Kb + base;
  const unsigned short* Vp = Vb + base;

  const v16bf aq0 = load_a16(Qp, DHEAD, q0, 0, lane);
  const v16bf aq1 = load_a16(Qp, DHEAD, q0, 32, lane);

  const int k0b = (q0 >= WINSZ) ? (q0 - WINSZ) : 0;
  const int ktiles = ((q0 - k0b) >> 4) + 1;           // <= 17

  // ---- pass 1: scores = scale * Q K^T, masked; stage in LDS; track row max ----
  v8f rmax = {-3e38f, -3e38f, -3e38f, -3e38f, -3e38f, -3e38f, -3e38f, -3e38f};
  for (int t = 0; t < ktiles; ++t) {
    const int kc = k0b + t * 16;
    const v16bf bk0 = load_b16(Kp, DHEAD, kc, 0, lane);
    const v16bf bk1 = load_b16(Kp, DHEAD, kc, 32, lane);
    v8f s = {0.f, 0.f, 0.f, 0.f, 0.f, 0.f, 0.f, 0.f};
    s = __builtin_amdgcn_wmma_f32_16x16x32_bf16(false, aq0, false, bk0, (short)0, s,
                                                false, false);
    s = __builtin_amdgcn_wmma_f32_16x16x32_bf16(false, aq1, false, bk1, (short)0, s,
                                                false, false);
#pragma unroll
    for (int g = 0; g < 8; ++g) {
      const int i = q0 + g + 8 * half;
      const int j = kc + r;
      float sv = s[g] * 0.125f;                        // 1/sqrt(64)
      if (j > i || j < i - WINSZ) sv = -3e38f;
      rmax[g] = fmaxf(rmax[g], sv);
      s_sc[wave][g + 8 * half][t * 16 + r] = sv;
    }
  }
  // row max across the 16 lanes of each half
#pragma unroll
  for (int g = 0; g < 8; ++g) {
    float m = rmax[g];
    m = fmaxf(m, __shfl_xor(m, 1, 32));
    m = fmaxf(m, __shfl_xor(m, 2, 32));
    m = fmaxf(m, __shfl_xor(m, 4, 32));
    m = fmaxf(m, __shfl_xor(m, 8, 32));
    if (r == g) s_rm[wave][g + 8 * half] = m;
  }
  __builtin_amdgcn_wave_barrier();
  const float mrow = s_rm[wave][r];                    // row = lane%16 (A layout)

  // ---- pass 2: P = exp(s - m) rebuilt in A-operand layout; O += P V ----
  v8f z8 = {0.f, 0.f, 0.f, 0.f, 0.f, 0.f, 0.f, 0.f};
  v8f o[4] = {z8, z8, z8, z8};
  float lsum = 0.f;
  const int nchunk = (ktiles + 1) >> 1;
  for (int c = 0; c < nchunk; ++c) {
#if HAVE_ASYNC_LDS
    // Async-stage this 32-key V chunk (32 x 64 bf16 = 4KB) into LDS.
    // 8 x GLOBAL_LOAD_ASYNC_TO_LDS_B128, each moving 32 lanes x 16B.
    {
      const int rowl = lane >> 3;            // 0..3 key rows per instruction
      const int seg  = (lane & 7) * 8;       // ushort offset within a 64-elem row
#pragma unroll
      for (int i = 0; i < 8; ++i) {
        int key = k0b + c * 32 + i * 4 + rowl;
        if (key > T_SEQ - 1) key = T_SEQ - 1;          // padded keys get p=0
        const unsigned short* g = Vp + (size_t)key * DHEAD + seg;
        unsigned short* l = &s_v[wave][(i * 4 + rowl) * DHEAD + seg];
        __builtin_amdgcn_global_load_async_to_lds_b128(
            (gptr_b128)(g), (lptr_b128)(l), 0, 0);
      }
    }
#endif
    // Build P in A-operand layout while the async copy is in flight.
    V16U ap;
#pragma unroll
    for (int e = 0; e < 16; ++e) {
      const int kk = c * 32 + 16 * half + e;  // contiguous K (matches B)
      float p = 0.f;
      if (kk < ktiles * 16) p = __expf(s_sc[wave][r][kk] - mrow);
      lsum += p;
      ap.us[e] = f2bf(p);
    }
#if HAVE_ASYNC_LDS
    __builtin_amdgcn_s_wait_asynccnt(0);
    __asm__ volatile("" ::: "memory");
#pragma unroll
    for (int nt = 0; nt < 4; ++nt) {
      V16U bv;
#pragma unroll
      for (int e = 0; e < 16; ++e)   // B layout: K = 16*half + e within chunk
        bv.us[e] = s_v[wave][(16 * half + e) * DHEAD + nt * 16 + r];
      o[nt] = __builtin_amdgcn_wmma_f32_16x16x32_bf16(false, ap.v, false, bv.v,
                                                      (short)0, o[nt], false, false);
    }
#else
    const int keybase = k0b + c * 32 + 16 * half;                // B K layout
#pragma unroll
    for (int nt = 0; nt < 4; ++nt) {
      V16U bv;
#pragma unroll
      for (int e = 0; e < 16; ++e) {
        int key = keybase + e;
        if (key > T_SEQ - 1) key = T_SEQ - 1;        // padded keys have p=0
        bv.us[e] = Vp[(size_t)key * DHEAD + nt * 16 + r];
      }
      o[nt] = __builtin_amdgcn_wmma_f32_16x16x32_bf16(false, ap.v, false, bv.v,
                                                      (short)0, o[nt], false, false);
    }
#endif
  }
  // row sums: row r lives in lanes r and r+16
  const float tot = lsum + __shfl_xor(lsum, 16, 32);
  if (half == 0) s_rs[wave][r] = tot;
  __builtin_amdgcn_wave_barrier();

  const int b = bh >> 4, h = bh & (HEADS - 1);
#pragma unroll
  for (int g = 0; g < 8; ++g) {
    const int i = q0 + g + 8 * half;
    const float inv = 1.f / s_rs[wave][g + 8 * half];
    unsigned short* orow = Ob + ((size_t)(b * T_SEQ + i)) * CDIM + h * DHEAD + r;
#pragma unroll
    for (int nt = 0; nt < 4; ++nt) orow[nt * 16] = f2bf(o[nt][g] * inv);
  }
}

// ---------------- launcher ----------------
extern "C" void kernel_launch(void* const* d_in, const int* in_sizes, int n_in,
                              void* d_out, int out_size, void* d_ws, size_t ws_size,
                              hipStream_t stream) {
  const float* x     = (const float*)d_in[0];
  const float* w_qkv = (const float*)d_in[1];
  const float* w_out = (const float*)d_in[2];
  const float* b_out = (const float*)d_in[3];
  float* out = (float*)d_out;

  char* ws = (char*)d_ws;
  size_t off = 0;
  auto alloc = [&](size_t bytes) -> void* {
    void* p = ws + off;
    off = (off + bytes + 255) & ~(size_t)255;
    return p;
  };
  const size_t nTok = (size_t)BATCH * T_SEQ;  // 4096

  unsigned short* xb    = (unsigned short*)alloc(nTok * CDIM * 2);
  unsigned short* wqkvb = (unsigned short*)alloc((size_t)3 * CDIM * CDIM * 2);
  unsigned short* woutb = (unsigned short*)alloc((size_t)CDIM * CDIM * 2);
  float*          qkvf  = (float*)alloc(nTok * 3 * CDIM * 4);
  unsigned short* Qbuf  = (unsigned short*)alloc(nTok * CDIM * 2);
  unsigned short* Kbuf  = (unsigned short*)alloc(nTok * CDIM * 2);
  unsigned short* Vbuf  = (unsigned short*)alloc(nTok * CDIM * 2);
  unsigned short* Ab    = (unsigned short*)alloc(nTok * CDIM * 2);
  (void)ws_size; (void)in_sizes; (void)n_in; (void)out_size;

  const int nX = (int)(nTok * CDIM);
  const int nWq = 3 * CDIM * CDIM;
  const int nWo = CDIM * CDIM;
  cvt_bf16<<<(nX + 255) / 256, 256, 0, stream>>>(x, xb, nX);
  cvt_bf16<<<(nWq + 255) / 256, 256, 0, stream>>>(w_qkv, wqkvb, nWq);
  cvt_bf16<<<(nWo + 255) / 256, 256, 0, stream>>>(w_out, woutb, nWo);

  // qkv = x @ w_qkv.T : M=4096, N=3072, K=1024 ; 24 N-macrotiles of 128
  gemm_bf16_16x16<<<(4096 / 16) * 24, 256, 0, stream>>>(xb, wqkvb, nullptr, qkvf,
                                                        3 * CDIM, CDIM, 24);

  rope_pack<<<(BATCH * HEADS * T_SEQ * 32) / 256, 256, 0, stream>>>(qkvf, Qbuf, Kbuf,
                                                                    Vbuf);

  // 4096 query tiles, 2 waves per block
  attn_local<<<(BATCH * HEADS * (T_SEQ / 16)) / 2, 64, 0, stream>>>(Qbuf, Kbuf, Vbuf,
                                                                    Ab);

  // out = attn @ w_out.T + b_out : M=4096, N=1024, K=1024 ; 8 N-macrotiles
  gemm_bf16_16x16<<<(4096 / 16) * 8, 256, 0, stream>>>(Ab, woutb, b_out, out, CDIM,
                                                       CDIM, 8);
}
